// tweet_model_22067541967831
// MI455X (gfx1250) — matrix operand order repaired and checked by reading, not compiled, not run on hardware
//
#include <hip/hip_runtime.h>
#include <stdint.h>

static constexpr int kH   = 2048;
static constexpr int kG4  = 8192;   // 4*kH gate rows
static constexpr int kE   = 64;
static constexpr int kV   = 32;
static constexpr int kT   = 290;
static constexpr int kSTART = 32;
static constexpr int kNWG = 64;
static constexpr int kTPB = 256;
static constexpr int kNT  = kNWG * kTPB;   // 16384 threads, 512 waves

typedef __attribute__((ext_vector_type(16))) __bf16 v16bf;
typedef __attribute__((ext_vector_type(8)))  __bf16 v8bf;
typedef __attribute__((ext_vector_type(8)))  float  v8f;

struct Params {
  const int*   tweet;
  const float* embed;
  const float* Wih0f; const float* Whh0f; const float* bih0; const float* bhh0;
  const float* Wih1f; const float* Whh1f; const float* bih1; const float* bhh1;
  const float* Wfcf;  const float* bfc;
  int* out;
  unsigned* cnt; unsigned* gen;
  float* gates0; float* gates1; float* bias0; float* bias1;
  __bf16* Wih0; __bf16* Whh0; __bf16* Wih1; __bf16* Whh1; __bf16* Wfc;
};

__device__ inline __bf16 f2bf(float f) {
  uint32_t u = __builtin_bit_cast(uint32_t, f);
  uint32_t r = (u + 0x7FFFu + ((u >> 16) & 1u)) >> 16;  // round-to-nearest-even
  unsigned short s = (unsigned short)r;
  return __builtin_bit_cast(__bf16, s);
}

// ---- device-wide sense-reversing barrier (agent scope) ----
__device__ inline void gbar(unsigned* cnt, unsigned* gen) {
  __syncthreads();
  __threadfence();   // release all global writes of this WG
  if (threadIdx.x == 0) {
    unsigned g = __hip_atomic_load(gen, __ATOMIC_RELAXED, __HIP_MEMORY_SCOPE_AGENT);
    unsigned a = __hip_atomic_fetch_add(cnt, 1u, __ATOMIC_ACQ_REL, __HIP_MEMORY_SCOPE_AGENT) + 1u;
    if (a == (unsigned)kNWG) {
      __hip_atomic_store(cnt, 0u, __ATOMIC_RELAXED, __HIP_MEMORY_SCOPE_AGENT);
      __hip_atomic_fetch_add(gen, 1u, __ATOMIC_RELEASE, __HIP_MEMORY_SCOPE_AGENT);
    } else {
      while (__hip_atomic_load(gen, __ATOMIC_RELAXED, __HIP_MEMORY_SCOPE_AGENT) == g)
        __builtin_amdgcn_s_sleep(2);
    }
  }
  __syncthreads();
  __threadfence();   // acquire: invalidate stale cached lines
}

// ---- JAX threefry2x32 block (20 rounds) ----
__device__ inline void tf2x32(uint32_t k0, uint32_t k1, uint32_t x0, uint32_t x1,
                              uint32_t* o0, uint32_t* o1) {
  uint32_t ks2 = 0x1BD11BDAu ^ k0 ^ k1;
  x0 += k0; x1 += k1;
#define TFR(r) { x0 += x1; x1 = (x1 << (r)) | (x1 >> (32 - (r))); x1 ^= x0; }
  TFR(13) TFR(15) TFR(26) TFR(6)  x0 += k1;  x1 += ks2 + 1u;
  TFR(17) TFR(29) TFR(16) TFR(24) x0 += ks2; x1 += k0  + 2u;
  TFR(13) TFR(15) TFR(26) TFR(6)  x0 += k0;  x1 += k1  + 3u;
  TFR(17) TFR(29) TFR(16) TFR(24) x0 += k1;  x1 += ks2 + 4u;
  TFR(13) TFR(15) TFR(26) TFR(6)  x0 += ks2; x1 += k0  + 5u;
#undef TFR
  *o0 = x0; *o1 = x1;
}

// A-fragment: vector `a` broadcast to all 16 rows of the 16x32 A tile.
// Lanes 0-15 hold K 0..15; lanes 16-31 hold K 16..31.
__device__ inline v16bf ldA(const __bf16* ap, int k0) {
  v8bf a0 = *(const v8bf*)(ap + k0);
  v8bf a1 = *(const v8bf*)(ap + k0 + 8);
  return __builtin_shufflevector(a0, a1, 0,1,2,3,4,5,6,7,8,9,10,11,12,13,14,15);
}
// B-fragment: 32x16 slice of W^T; lane l<16 -> col row0+l, K k0..k0+15 (contiguous in W row);
// lane l>=16 -> col row0+l-16, K k0+16..k0+31.
__device__ inline v16bf ldB(const __bf16* wp, int k0) {
  v8bf b0 = *(const v8bf*)(wp + k0);
  v8bf b1 = *(const v8bf*)(wp + k0 + 8);
  return __builtin_shufflevector(b0, b1, 0,1,2,3,4,5,6,7,8,9,10,11,12,13,14,15);
}

struct Acc2 { v8f a, b; };

// gates[row0 .. row0+15] += W[row0:row0+16, :] @ x   (K multiple of 64)
__device__ inline Acc2 mv(const __bf16* __restrict__ W, int K, const __bf16* x,
                          int row0, int lane, Acc2 acc) {
  const int m  = lane & 15;
  const int hi = lane >> 4;
  const __bf16* wp = W + (size_t)(row0 + m) * (size_t)K + hi * 16;
  const __bf16* ap = x + hi * 16;
  for (int k0 = 0; k0 < K; k0 += 64) {
    v16bf A0 = ldA(ap, k0);
    v16bf B0 = ldB(wp, k0);
    acc.a = __builtin_amdgcn_wmma_f32_16x16x32_bf16(false, A0, false, B0, (short)0, acc.a, false, false);
    v16bf A1 = ldA(ap, k0 + 32);
    v16bf B1 = ldB(wp, k0 + 32);
    acc.b = __builtin_amdgcn_wmma_f32_16x16x32_bf16(false, A1, false, B1, (short)0, acc.b, false, false);
  }
  return acc;
}

__device__ inline Acc2 accz() {
  Acc2 r; r.a = (v8f){0,0,0,0,0,0,0,0}; r.b = (v8f){0,0,0,0,0,0,0,0}; return r;
}

__device__ inline void cvt4(const float* __restrict__ s, __bf16* __restrict__ d,
                            int n, int g) {
  for (int i = g * 4; i < n; i += kNT * 4) {
    float4 v = *(const float4*)(s + i);
    d[i]     = f2bf(v.x);
    d[i + 1] = f2bf(v.y);
    d[i + 2] = f2bf(v.z);
    d[i + 3] = f2bf(v.w);
  }
}

__global__ void k_init(Params P) {
  int t = threadIdx.x;
  if (t == 0) { *P.cnt = 0u; *P.gen = 0u; }
  if (t < kSTART) P.out[t] = P.tweet[t];
}

__global__ __launch_bounds__(kTPB) void k_lstm(Params P) {
  __shared__ __align__(16) __bf16 sh_h0[kH];
  __shared__ __align__(16) __bf16 sh_h1[kH];
  __shared__ float sh_c0[kH];
  __shared__ float sh_c1[kH];
  __shared__ __align__(16) __bf16 sh_x[kE];
  __shared__ float sh_logits[kV];
  __shared__ float sh_val[kV];
  __shared__ int   sh_tok;

  const int tid  = threadIdx.x;
  const int gtid = blockIdx.x * kTPB + tid;
  const int lane = tid & 31;
  const int gw16 = (gtid >> 5) * 16;   // global wave -> 16-row gate tile
  const int lw   = tid >> 5;           // local wave in WG

  // ---- once per launch: fp32 -> bf16 weights, fused biases ----
  cvt4(P.Wih0f, P.Wih0, kG4 * kE, gtid);
  cvt4(P.Whh0f, P.Whh0, kG4 * kH, gtid);
  cvt4(P.Wih1f, P.Wih1, kG4 * kH, gtid);
  cvt4(P.Whh1f, P.Whh1, kG4 * kH, gtid);
  cvt4(P.Wfcf,  P.Wfc,  kV  * kH, gtid);
  for (int j = gtid; j < kG4; j += kNT) {
    P.bias0[j] = P.bih0[j] + P.bhh0[j];
    P.bias1[j] = P.bih1[j] + P.bhh1[j];
  }
  const __bf16 bz = __builtin_bit_cast(__bf16, (unsigned short)0);
  for (int j = tid; j < kH; j += kTPB) {
    sh_c0[j] = 0.f; sh_c1[j] = 0.f; sh_h0[j] = bz; sh_h1[j] = bz;
  }
  if (tid < kE) {
    int tok0 = P.tweet[0];
    sh_x[tid] = f2bf(P.embed[tok0 * kE + tid]);
  }
  gbar(P.cnt, P.gen);

  for (int t = 0; t < kT; ++t) {
    // ---- B: gates0 = Wih0 @ x + Whh0 @ h0 (distributed: 512 waves x 16 rows)
    {
      Acc2 acc = accz();
      acc = mv(P.Wih0, kE, sh_x,  gw16, lane, acc);
      acc = mv(P.Whh0, kH, sh_h0, gw16, lane, acc);
      if (lane < 16) P.gates0[gw16 + lane] = acc.a[0] + acc.b[0];
    }
    gbar(P.cnt, P.gen);

    // ---- C: layer-0 cell update (redundant per WG, state in LDS)
    for (int j = tid; j < kH; j += kTPB) {
      float iv = P.gates0[j]          + P.bias0[j];
      float fv = P.gates0[j + kH]     + P.bias0[j + kH];
      float gv = P.gates0[j + 2*kH]   + P.bias0[j + 2*kH];
      float ov = P.gates0[j + 3*kH]   + P.bias0[j + 3*kH];
      iv = 1.f / (1.f + expf(-iv));
      fv = 1.f / (1.f + expf(-fv));
      ov = 1.f / (1.f + expf(-ov));
      gv = tanhf(gv);
      float cn = fv * sh_c0[j] + iv * gv;
      sh_c0[j] = cn;
      sh_h0[j] = f2bf(ov * tanhf(cn));
    }
    __syncthreads();

    // ---- D: gates1 = Wih1 @ h0 + Whh1 @ h1 (distributed)
    {
      Acc2 acc = accz();
      acc = mv(P.Wih1, kH, sh_h0, gw16, lane, acc);
      acc = mv(P.Whh1, kH, sh_h1, gw16, lane, acc);
      if (lane < 16) P.gates1[gw16 + lane] = acc.a[0] + acc.b[0];
    }
    gbar(P.cnt, P.gen);

    // ---- E: layer-1 cell update (redundant per WG)
    for (int j = tid; j < kH; j += kTPB) {
      float iv = P.gates1[j]          + P.bias1[j];
      float fv = P.gates1[j + kH]     + P.bias1[j + kH];
      float gv = P.gates1[j + 2*kH]   + P.bias1[j + 2*kH];
      float ov = P.gates1[j + 3*kH]   + P.bias1[j + 3*kH];
      iv = 1.f / (1.f + expf(-iv));
      fv = 1.f / (1.f + expf(-fv));
      ov = 1.f / (1.f + expf(-ov));
      gv = tanhf(gv);
      float cn = fv * sh_c1[j] + iv * gv;
      sh_c1[j] = cn;
      sh_h1[j] = f2bf(ov * tanhf(cn));
    }
    __syncthreads();

    // ---- F: logits = Wfc @ h1 + b_fc (local waves 0,1; redundant per WG)
    if (lw < 2) {
      Acc2 acc = accz();
      acc = mv(P.Wfc, kH, sh_h1, lw * 16, lane, acc);
      if (lane < 16) {
        int r = lw * 16 + lane;
        sh_logits[r] = acc.a[0] + acc.b[0] + P.bfc[r];
      }
    }
    __syncthreads();

    // ---- A': pick token for step t+1 and fetch its embedding
    int nt = t + 1;
    if (nt < kT) {
      int tokn;
      if (nt >= kSTART) {
        if (tid < kV) {
          uint32_t fk0, fk1, r0, r1;
          tf2x32(0u, 42u, 0u, (uint32_t)(nt - kSTART), &fk0, &fk1);   // fold_in(key(42), i)
          uint32_t c0 = (uint32_t)(tid & 15), c1 = c0 + 16u;
          tf2x32(fk0, fk1, c0, c1, &r0, &r1);
          uint32_t bits = (tid < 16) ? r0 : r1;
          float f = __builtin_bit_cast(float, (bits >> 9) | 0x3f800000u) - 1.0f;
          float u = fmaxf(f, 1.17549435e-38f);
          float gum = -logf(-logf(u));
          sh_val[tid] = sh_logits[tid] / 0.2f + gum;
        }
        __syncthreads();
        if (tid == 0) {
          float best = sh_val[0]; int bi = 0;
          for (int k = 1; k < kV; ++k)
            if (sh_val[k] > best) { best = sh_val[k]; bi = k; }
          sh_tok = bi;
          if (blockIdx.x == 0) P.out[nt] = bi;
        }
        __syncthreads();
        tokn = sh_tok;
      } else {
        tokn = P.tweet[nt];
      }
      if (tid < kE) sh_x[tid] = f2bf(P.embed[tokn * kE + tid]);
    }
    __syncthreads();
  }
}

extern "C" void kernel_launch(void* const* d_in, const int* in_sizes, int n_in,
                              void* d_out, int out_size, void* d_ws, size_t ws_size,
                              hipStream_t stream) {
  (void)in_sizes; (void)n_in; (void)out_size; (void)ws_size;
  Params P;
  P.tweet = (const int*)d_in[0];
  P.embed = (const float*)d_in[1];
  P.Wih0f = (const float*)d_in[2];
  P.Whh0f = (const float*)d_in[3];
  P.bih0  = (const float*)d_in[4];
  P.bhh0  = (const float*)d_in[5];
  P.Wih1f = (const float*)d_in[6];
  P.Whh1f = (const float*)d_in[7];
  P.bih1  = (const float*)d_in[8];
  P.bhh1  = (const float*)d_in[9];
  P.Wfcf  = (const float*)d_in[10];
  P.bfc   = (const float*)d_in[11];
  P.out   = (int*)d_out;

  char* w = (char*)d_ws;
  size_t off = 0;
  auto take = [&](size_t bytes) -> void* {
    void* r = w + off;
    off = (off + bytes + 255) & ~(size_t)255;
    return r;
  };
  P.cnt    = (unsigned*)take(4);
  P.gen    = (unsigned*)take(4);
  P.gates0 = (float*)take(sizeof(float) * kG4);
  P.gates1 = (float*)take(sizeof(float) * kG4);
  P.bias0  = (float*)take(sizeof(float) * kG4);
  P.bias1  = (float*)take(sizeof(float) * kG4);
  P.Wih0   = (__bf16*)take(2ull * kG4 * kE);
  P.Whh0   = (__bf16*)take(2ull * kG4 * kH);
  P.Wih1   = (__bf16*)take(2ull * kG4 * kH);
  P.Whh1   = (__bf16*)take(2ull * kG4 * kH);
  P.Wfc    = (__bf16*)take(2ull * kV * kH);
  // total workspace ~97.4 MB (fits device L2 for the hot weight arrays)

  k_init<<<1, 64, 0, stream>>>(P);
  k_lstm<<<kNWG, kTPB, 0, stream>>>(P);
}
